// Model_2027224564353
// MI455X (gfx1250) — matrix operand order repaired and checked by reading, not compiled, not run on hardware
//
#include <hip/hip_runtime.h>
#include <hip/hip_bf16.h>
#include <math.h>

typedef __attribute__((ext_vector_type(16))) __bf16 v16bf;
typedef __attribute__((ext_vector_type(8)))  float  v8f;

#define B_SZ   8
#define SEQ    768
#define ENC    21
#define PRED   192
#define NH     256
#define MODES  32
#define BE     (B_SZ*ENC)   /* 168 */
#define TM     11           /* ceil(168/16) M tiles */
#define TK2    512          /* GEMM2 K = 64*256/32  */
#define TK3    8            /* GEMM3 K = 256/32     */
#define NCOMBO 64           /* 32 modes x {re,im}   */
#define TKMAX  24           /* 768/32               */
#define LMAX   768
#define B1ELEM ((size_t)16*TKMAX*512)  /* per-scale B1 elements */
#define B3ELEM ((size_t)12*TK3*512)    /* per-scale B3 elements */
#define PI_F   3.14159265358979323846f

/* ---------------- Stage 1: RevIN normalization ---------------- */
__global__ void norm_kernel(const float* __restrict__ x, const float* __restrict__ aw,
                            const float* __restrict__ ab, float* __restrict__ xc,
                            float* __restrict__ means, float* __restrict__ stdev) {
  __shared__ float red[256];
  int be = blockIdx.x, b = be / ENC, e = be % ENC, tid = threadIdx.x;
  const float* xb = x + (size_t)b * SEQ * ENC + e;
  float s = 0.f;
  for (int t = tid; t < SEQ; t += 256) s += xb[(size_t)t * ENC];
  red[tid] = s; __syncthreads();
  for (int w = 128; w > 0; w >>= 1) { if (tid < w) red[tid] += red[tid + w]; __syncthreads(); }
  float mean = red[0] / (float)SEQ; __syncthreads();
  float v = 0.f;
  for (int t = tid; t < SEQ; t += 256) { float d = xb[(size_t)t*ENC] - mean; v += d*d; }
  red[tid] = v; __syncthreads();
  for (int w = 128; w > 0; w >>= 1) { if (tid < w) red[tid] += red[tid + w]; __syncthreads(); }
  float sd = sqrtf(red[0] / (float)SEQ + 1e-5f);
  if (tid == 0) { means[be] = mean; stdev[be] = sd; }
  float ww = aw[e], bb = ab[e];
  for (int t = tid; t < SEQ; t += 256)
    xc[(size_t)b*SEQ*ENC + (size_t)t*ENC + e] = ((xb[(size_t)t*ENC] - mean) / sd) * ww + bb;
}

/* ------- Stage 2: HiPPO bilinear discretization, all 3 scales in parallel.
   f64 Gauss-Jordan, 1024 threads = 4 column-lanes per matrix row. ------- */
__global__ void hippo_gj_kernel(double* __restrict__ GJ, float* __restrict__ Ad,
                                float* __restrict__ Bd) {
  const int COLS = 2*NH + 1;
  int s = blockIdx.x;
  double dt = 1.0 / (double)(PRED << s);
  double* Gm  = GJ + (size_t)s * NH * COLS;
  float*  Ads = Ad + (size_t)s * NH * NH;
  float*  Bds = Bd + (size_t)s * NH;
  int tid = threadIdx.x;
  int r = tid >> 2, q = tid & 3;
  double Rr = 2.0*r + 1.0;
  for (int c = q; c < NH; c += 4) {
    double a = (r < c) ? -Rr : (((r - c + 1) & 1) ? -Rr : Rr);
    double idn = (r == c) ? 1.0 : 0.0;
    Gm[(size_t)r*COLS + c]      = idn - 0.5*dt*a;   /* M = I - dt/2 A */
    Gm[(size_t)r*COLS + NH + c] = idn + 0.5*dt*a;   /* N = I + dt/2 A */
  }
  if (q == 0) Gm[(size_t)r*COLS + 2*NH] = dt * ((r & 1) ? -Rr : Rr);  /* dt*Bv */
  __syncthreads();
  for (int p = 0; p < NH; ++p) {
    double piv = Gm[(size_t)p*COLS + p];
    __syncthreads();
    for (int c = p + tid; c < COLS; c += 1024) Gm[(size_t)p*COLS + c] /= piv;
    __syncthreads();
    if (r != p) {
      double f = Gm[(size_t)r*COLS + p];
      if (f != 0.0)
        for (int c = p + 1 + q; c < COLS; c += 4)
          Gm[(size_t)r*COLS + c] -= f * Gm[(size_t)p*COLS + c];
      /* col p is never read again: no zeroing needed */
    }
    __syncthreads();
  }
  for (int c = q; c < NH; c += 4) Ads[r*NH + c] = (float)Gm[(size_t)r*COLS + NH + c];
  if (q == 0) Bds[r] = (float)Gm[(size_t)r*COLS + 2*NH];
}

/* ------- Stage 3: Legendre eval matrix (3 scales parallel), GEMM3 B-operand ------- */
__device__ __forceinline__ void store_b3(__bf16* B3, int o, int p, float v) {
  int tn = p >> 4, tk = o >> 5, kk = o & 31;
  int lane = (p & 15) + (((kk >> 3) & 1) << 4), e = (kk & 7) + ((kk >> 4) << 3);
  B3[(((size_t)tn*TK3 + tk)*32 + lane)*16 + e] = (__bf16)v;
}
__global__ void eval_pack_kernel(__bf16* __restrict__ B3all) {
  int s = blockIdx.x;
  int L = PRED << s;
  double dt = 1.0 / (double)L;
  __bf16* B3 = B3all + (size_t)s * B3ELEM;
  int p = threadIdx.x;
  if (p >= PRED) return;
  int row = L - PRED + p;
  double xv = 1.0 - 2.0 * ((double)row * dt);
  store_b3(B3, 0, p, 1.0f);
  store_b3(B3, 1, p, (float)xv);
  double pm1 = 1.0, pc = xv;
  for (int o = 2; o < NH; ++o) {
    int n = o - 1;
    double pn = ((2.0*n + 1.0)*xv*pc - (double)n*pm1) / (double)(n + 1);
    store_b3(B3, o, p, (float)pn);
    pm1 = pc; pc = pn;
  }
}

/* ------- Stage 4: K chain, log-depth.
   Seed K_0..K_31 serially (3 scales parallel), then doubling with Ad^(32*2^t). ---- */
__global__ void kchain_seed_kernel(const float* __restrict__ Ad, const float* __restrict__ Bd,
                                   float* __restrict__ Kflat) {
  int s = blockIdx.x;
  __shared__ float Kc[NH];
  int i = threadIdx.x;
  const float* row = Ad + (size_t)s*NH*NH + (size_t)i*NH;
  float* Ks = Kflat + (size_t)s*LMAX*NH;
  Kc[i] = Bd[s*NH + i];
  __syncthreads();
  for (int j = 0; j < 32; ++j) {
    Ks[(size_t)j*NH + i] = Kc[i];
    if (j < 31) {
      float acc = 0.f;
      for (int p = 0; p < NH; ++p) acc += row[p] * Kc[p];
      __syncthreads();
      Kc[i] = acc;
      __syncthreads();
    }
  }
}

/* C_s = A_s x A_s (256x256 f32 squaring), batched over 3 scales */
__global__ void mat_sq_kernel(const float* __restrict__ src, float* __restrict__ dst) {
  int s = blockIdx.z;
  const float* A = src + (size_t)s*NH*NH;
  float* C = dst + (size_t)s*NH*NH;
  int col = blockIdx.x*16 + threadIdx.x;
  int row = blockIdx.y*16 + threadIdx.y;
  float acc = 0.f;
  for (int p = 0; p < NH; ++p) acc += A[(size_t)row*NH + p] * A[(size_t)p*NH + col];
  C[(size_t)row*NH + col] = acc;
}

/* K_{32*nb + r} = Ad^{32*nb} * K_r  for r in [0, 32*add);  R holds Ad^{32*2^t} */
__global__ void kchain_double_kernel(float* __restrict__ Kflat, const float* __restrict__ R,
                                     int t) {
  int s = blockIdx.z;
  int TKs = 6 << s;
  int nb = 1 << t;
  if (nb >= TKs) return;
  int add = (nb < TKs - nb) ? nb : (TKs - nb);
  int rows = 32 * add;
  int r = blockIdx.y*16 + threadIdx.y;
  int i = blockIdx.x*16 + threadIdx.x;
  if (r >= rows) return;
  const float* Ks = Kflat + (size_t)s*LMAX*NH;
  const float* Rs = R + (size_t)s*NH*NH;
  float acc = 0.f;
  for (int p = 0; p < NH; ++p) acc += Ks[(size_t)r*NH + p] * Rs[(size_t)i*NH + p];
  Kflat[(size_t)s*LMAX*NH + (size_t)(32*nb + r)*NH + i] = acc;
}

/* pack f32 K (L x 256) into WMMA B-operand layout (bf16), 3 scales batched */
__global__ void pack_b1_kernel(const float* __restrict__ Kflat, __bf16* __restrict__ B1all) {
  int s = blockIdx.z;
  int L = PRED << s, TK = L / 32;
  int idx = blockIdx.x*256 + threadIdx.x;
  int i = idx & 255, j = idx >> 8;
  if (j >= L) return;
  float v = Kflat[(size_t)s*LMAX*NH + (size_t)j*NH + i];
  int tn = i >> 4, tk = j >> 5, kk = j & 31;
  int lane = (i & 15) + (((kk >> 3) & 1) << 4), e = (kk & 7) + ((kk >> 4) << 3);
  B1all[(size_t)s*B1ELEM + (((size_t)tn*TK + tk)*32 + lane)*16 + e] = (__bf16)v;
}

/* ------- Stage 5: prefix-DFT G matrix, packed as GEMM1 A-operand -------
   a_k = sum_j G_k[.,j] K_j ; G_k[j] = e^{-i w j} * S_k[L-1-j]            */
__global__ void prefix_g_kernel(const float* __restrict__ xc, __bf16* __restrict__ Gz,
                                int L, int numTK) {
  int be = blockIdx.x, k = threadIdx.x;
  int b = be / ENC, e = be % ENC;
  const float* f = xc + (size_t)b*SEQ*ENC + (size_t)(SEQ - L)*ENC + e;
  float w = 2.f * PI_F * (float)k / (float)L;
  float Sr = 0.f, Si = 0.f;
  int tm = be >> 4, mr = be & 15;
  size_t tilesz = (size_t)TM * numTK * 512;
  __bf16* Gr_base = Gz + (size_t)(2*k)     * tilesz;
  __bf16* Gi_base = Gz + (size_t)(2*k + 1) * tilesz;
  for (int t = 0; t < L; ++t) {
    float ft = f[(size_t)t*ENC];
    float sa, ca; __sincosf(w * (float)t, &sa, &ca);
    Sr += ft * ca; Si -= ft * sa;
    int j = L - 1 - t;
    float sj, cj; __sincosf(w * (float)j, &sj, &cj);
    float Gr = cj*Sr + sj*Si;
    float Gi = cj*Si - sj*Sr;
    int kk = j & 31, tk = j >> 5;
    int h2 = (kk >> 3) & 1;
    int e2 = (kk & 7) + ((kk >> 4) << 3);
    int lane = mr + (h2 << 4);
    size_t idx = (((size_t)tm*numTK + tk)*32 + lane)*16 + e2;
    Gr_base[idx] = (__bf16)Gr;
    Gi_base[idx] = (__bf16)Gi;
  }
}

/* ------- Stage 6: spectral weights with folded irfft phase, GEMM2 B ------- */
__global__ void cprime_pack_kernel(const float* __restrict__ wr, const float* __restrict__ wi,
                                   __bf16* __restrict__ B2, int L, int scale) {
  size_t idx = (size_t)blockIdx.x * blockDim.x + threadIdx.x;
  if (idx >= (size_t)NCOMBO*NH*NH) return;
  int o = (int)(idx % NH);
  int i = (int)((idx / NH) % NH);
  int combo = (int)(idx / ((size_t)NH*NH));
  int k = combo >> 1, c = combo & 1;
  size_t widx = (((size_t)scale*NH + i)*NH + o)*MODES + k;
  float wrv = wr[widx], wiv = wi[widx];
  float phi = 2.f * PI_F * (float)k * (float)(PRED - 1) / (float)L;
  float sp, cp; __sincosf(phi, &sp, &cp);
  float rho = ((k == 0) ? 1.f : 2.f) / (float)L;
  float val = (c == 0) ? rho * (wrv*cp - wiv*sp) : -rho * (wrv*sp + wiv*cp);
  int r2 = combo * NH + i;
  int tk = r2 >> 5, kk = r2 & 31;
  int tn = o >> 4, lane = (o & 15) + (((kk >> 3) & 1) << 4), e = (kk & 7) + ((kk >> 4) << 3);
  B2[(((size_t)tn*TK2 + tk)*32 + lane)*16 + e] = (__bf16)val;
}

/* ---------------- WMMA tile GEMM core (bf16 -> f32) ---------------- */
__device__ __forceinline__ v8f wmma_loop(const __bf16* __restrict__ A, const __bf16* __restrict__ B,
                                         int tm, int tn, int numTK, int lane) {
  v8f acc = {};
  const v16bf* Av = (const v16bf*)A;
  const v16bf* Bv = (const v16bf*)B;
  size_t aoff = (size_t)tm * numTK * 32 + lane;
  size_t boff = (size_t)tn * numTK * 32 + lane;
  for (int tk = 0; tk < numTK; ++tk) {
    v16bf a = Av[aoff + (size_t)tk * 32];
    v16bf b = Bv[boff + (size_t)tk * 32];
    acc = __builtin_amdgcn_wmma_f32_16x16x32_bf16(false, a, false, b, (short)0, acc, false, false);
  }
  return acc;
}

/* GEMM1: a[combo][be][i] = G_combo (168xL) x K (Lx256); epilogue packs GEMM2 A */
__global__ void gemm1_kernel(const __bf16* __restrict__ Gz, const __bf16* __restrict__ B1,
                             __bf16* __restrict__ A2, int numTK) {
  int lane = threadIdx.x, tm = blockIdx.x, tn = blockIdx.y, combo = blockIdx.z;
  const __bf16* A = Gz + (size_t)combo * TM * numTK * 512;
  v8f acc = wmma_loop(A, B1, tm, tn, numTK, lane);
  int n = lane & 15, h = lane >> 4;
  int i = tn * 16 + n;
  int r2 = combo * NH + i;
  int tk2 = r2 >> 5, kk = r2 & 31;
  int h2 = (kk >> 3) & 1;
  int e2 = (kk & 7) + ((kk >> 4) << 3);
  for (int r = 0; r < 8; ++r) {
    int mr = r + (h << 3);
    int lane2 = mr + (h2 << 4);
    A2[(((size_t)tm * TK2 + tk2) * 32 + lane2) * 16 + e2] = (__bf16)acc[r];
  }
}

/* GEMM2: out1[be][o] = a (168x16384) x C' (16384x256); epilogue packs GEMM3 A */
__global__ void gemm2_kernel(const __bf16* __restrict__ A2, const __bf16* __restrict__ B2,
                             __bf16* __restrict__ A3) {
  int lane = threadIdx.x, tm = blockIdx.x, tn = blockIdx.y;
  v8f acc = wmma_loop(A2, B2, tm, tn, TK2, lane);
  int n = lane & 15, h = lane >> 4;
  int o = tn * 16 + n;
  int tk3 = o >> 5, kk = o & 31;
  int h2 = (kk >> 3) & 1;
  int e2 = (kk & 7) + ((kk >> 4) << 3);
  for (int r = 0; r < 8; ++r) {
    int mr = r + (h << 3);
    int lane2 = mr + (h2 << 4);
    A3[(((size_t)tm * TK3 + tk3) * 32 + lane2) * 16 + e2] = (__bf16)acc[r];
  }
}

/* GEMM3: xdec[be][p] = out1 (168x256) x evT (256x192) */
__global__ void gemm3_kernel(const __bf16* __restrict__ A3, const __bf16* __restrict__ B3,
                             float* __restrict__ xdec) {
  int lane = threadIdx.x, tm = blockIdx.x, tn = blockIdx.y;
  v8f acc = wmma_loop(A3, B3, tm, tn, TK3, lane);
  int n = lane & 15, h = lane >> 4;
  int p = tn * 16 + n;
  for (int r = 0; r < 8; ++r) {
    int be = tm * 16 + r + (h << 3);
    if (be < BE) xdec[(size_t)be * PRED + p] = acc[r];
  }
}

/* ---------------- Stage 7: 3-scale MLP mix + de-normalization ---------------- */
__global__ void final_kernel(const float* __restrict__ xdec, const float* __restrict__ mlp_w,
                             const float* __restrict__ mlp_b, const float* __restrict__ aw,
                             const float* __restrict__ ab, const float* __restrict__ means,
                             const float* __restrict__ stdev, float* __restrict__ out) {
  int idx = blockIdx.x * blockDim.x + threadIdx.x;
  if (idx >= B_SZ*PRED*ENC) return;
  int e = idx % ENC;
  int p = (idx / ENC) % PRED;
  int b = idx / (ENC*PRED);
  int be = b*ENC + e;
  float v = mlp_b[0];
  for (int s = 0; s < 3; ++s)
    v += xdec[(size_t)s*BE*PRED + (size_t)be*PRED + p] * mlp_w[s];
  float xd = (v - ab[e]) / (aw[e] + 1e-10f);
  out[idx] = xd * stdev[be] + means[be];
}

extern "C" void kernel_launch(void* const* d_in, const int* in_sizes, int n_in,
                              void* d_out, int out_size, void* d_ws, size_t ws_size,
                              hipStream_t stream) {
  (void)in_sizes; (void)n_in; (void)out_size; (void)ws_size;
  const float* x_enc   = (const float*)d_in[0];
  const float* aw      = (const float*)d_in[4];
  const float* ab      = (const float*)d_in[5];
  const float* spec_wr = (const float*)d_in[6];
  const float* spec_wi = (const float*)d_in[7];
  const float* mlp_w   = (const float*)d_in[8];
  const float* mlp_b   = (const float*)d_in[9];
  float* out = (float*)d_out;

  char* ws = (char*)d_ws;
  size_t off = 0;
  auto alloc = [&](size_t bytes) -> char* {
    char* p = ws + off;
    off = (off + bytes + 255) & ~(size_t)255;
    return p;
  };
  float*  xc    = (float*) alloc((size_t)B_SZ*SEQ*ENC*4);
  float*  means = (float*) alloc(BE*4);
  float*  stdev = (float*) alloc(BE*4);
  double* GJ    = (double*)alloc((size_t)3*NH*(2*NH+1)*8);
  float*  Ad    = (float*) alloc((size_t)3*NH*NH*4);
  float*  Bd    = (float*) alloc((size_t)3*NH*4);
  float*  R0    = (float*) alloc((size_t)3*NH*NH*4);
  float*  R1    = (float*) alloc((size_t)3*NH*NH*4);
  float*  Kflat = (float*) alloc((size_t)3*LMAX*NH*4);
  __bf16* B1    = (__bf16*)alloc((size_t)3*B1ELEM*2);
  __bf16* B3    = (__bf16*)alloc((size_t)3*B3ELEM*2);
  __bf16* Gz    = (__bf16*)alloc((size_t)NCOMBO*TM*TKMAX*512*2);
  __bf16* A2    = (__bf16*)alloc((size_t)TM*TK2*512*2);
  __bf16* B2    = (__bf16*)alloc((size_t)16*TK2*512*2);
  __bf16* A3    = (__bf16*)alloc((size_t)TM*TK3*512*2);
  float*  xdec  = (float*) alloc((size_t)3*BE*PRED*4);

  /* input normalization + all-scale constant setup (batched, log-depth) */
  norm_kernel<<<BE, 256, 0, stream>>>(x_enc, aw, ab, xc, means, stdev);
  hippo_gj_kernel<<<3, 1024, 0, stream>>>(GJ, Ad, Bd);
  eval_pack_kernel<<<3, 256, 0, stream>>>(B3);
  kchain_seed_kernel<<<3, 256, 0, stream>>>(Ad, Bd, Kflat);

  dim3 sqg(NH/16, NH/16, 3), sqb(16, 16);
  mat_sq_kernel<<<sqg, sqb, 0, stream>>>(Ad, R0);   /* Ad^2  */
  mat_sq_kernel<<<sqg, sqb, 0, stream>>>(R0, R1);   /* Ad^4  */
  mat_sq_kernel<<<sqg, sqb, 0, stream>>>(R1, R0);   /* Ad^8  */
  mat_sq_kernel<<<sqg, sqb, 0, stream>>>(R0, R1);   /* Ad^16 */
  mat_sq_kernel<<<sqg, sqb, 0, stream>>>(R1, R0);   /* Ad^32 */
  float* Rcur = R0; float* Ralt = R1;
  for (int t = 0; t < 5; ++t) {
    kchain_double_kernel<<<dim3(NH/16, 2 << t, 3), sqb, 0, stream>>>(Kflat, Rcur, t);
    if (t < 4) {
      mat_sq_kernel<<<sqg, sqb, 0, stream>>>(Rcur, Ralt);
      float* tmp = Rcur; Rcur = Ralt; Ralt = tmp;
    }
  }
  pack_b1_kernel<<<dim3(LMAX*NH/256, 1, 3), 256, 0, stream>>>(Kflat, B1);

  /* per-scale WMMA pipeline */
  for (int s = 0; s < 3; ++s) {
    int L = PRED << s;
    int TK = L / 32;
    hipMemsetAsync(Gz, 0, (size_t)NCOMBO*TM*TK*512*2, stream);
    prefix_g_kernel<<<BE, 32, 0, stream>>>(xc, Gz, L, TK);
    cprime_pack_kernel<<<(NCOMBO*NH*NH + 255)/256, 256, 0, stream>>>(spec_wr, spec_wi, B2, L, s);
    gemm1_kernel<<<dim3(TM, NH/16, NCOMBO), 32, 0, stream>>>(Gz, B1 + (size_t)s*B1ELEM, A2, TK);
    gemm2_kernel<<<dim3(TM, NH/16), 32, 0, stream>>>(A2, B2, A3);
    gemm3_kernel<<<dim3(TM, PRED/16), 32, 0, stream>>>(A3, B3 + (size_t)s*B3ELEM,
                                                       xdec + (size_t)s*BE*PRED);
  }

  final_kernel<<<(B_SZ*PRED*ENC + 255)/256, 256, 0, stream>>>(
      xdec, mlp_w, mlp_b, aw, ab, means, stdev, out);
}
